// LSTM_60284160967156
// MI455X (gfx1250) — matrix operand order repaired and checked by reading, not compiled, not run on hardware
//
#include <hip/hip_runtime.h>
#include <hip/hip_bf16.h>

// LSTM forward on gfx1250 (MI455X), bf16 WMMA with f32 accumulation.
// Phase 1: xWx = x @ Wx + b  (one large WMMA GEMM, M=N*T=32768, K=512, C=2048)
// Phase 2: 512 sequential fused step kernels: a = xWx[:,t,:] + h@Wh -> gates -> c,h
// Both WMMA loops are software-pipelined (double-buffered fragments) so the
// compiler can issue partial s_wait_loadcnt instead of full drains before WMMA.
// Workspace layout (~306 MB): xWx f32 | x bf16 | Wx packed | Wh packed | h ping/pong | c

#define NB 64
#define TT 512
#define DD 512
#define HH 512
#define FOURH 2048

typedef __attribute__((ext_vector_type(16))) __bf16 v16bf;
typedef __attribute__((ext_vector_type(8)))  __bf16 v8bf;
typedef __attribute__((ext_vector_type(8)))  float  v8f;

__device__ __forceinline__ __bf16 f2bf(float f) {
  // round-to-nearest-even f32 -> bf16
  unsigned u = __float_as_uint(f);
  u = (u + 0x7FFFu + ((u >> 16) & 1u)) >> 16;
  unsigned short s = (unsigned short)u;
  __bf16 b;
  __builtin_memcpy(&b, &s, sizeof(s));
  return b;
}

// Build a 16x32 (MxK) bf16 A-fragment for one lane from a row-major matrix.
// ISA layout: lanes 0-15 hold K={0..7,16..23}, lanes 16-31 hold K={8..15,24..31};
// caller passes p = row_ptr + k0 + (lane>=16 ? 8 : 0).
__device__ __forceinline__ v16bf load_a_frag(const __bf16* __restrict__ p) {
  v8bf lo = *(const v8bf*)(p);
  v8bf hi = *(const v8bf*)(p + 16);
  v16bf a;
#pragma unroll
  for (int i = 0; i < 8; ++i) { a[i] = lo[i]; a[i + 8] = hi[i]; }
  return a;
}

// Pack a K x C row-major f32 matrix into bf16 WMMA B-fragment order:
// P[((nt*numKt + kt)*32 + lane)*16 + h], where element h of lane holds
// W[kt*32 + (lane>=16 ? 8 : 0) + (h<8 ? h : h+8)][nt*16 + (lane&15)].
__global__ void pack_b_kernel(const float* __restrict__ W, __bf16* __restrict__ P,
                              int K, int C) {
  int idx = blockIdx.x * blockDim.x + threadIdx.x;
  if (idx >= K * C) return;
  int h    = idx & 15;
  int lane = (idx >> 4) & 31;
  int rest = idx >> 9;            // nt*numKt + kt
  int numKt = K >> 5;
  int kt = rest % numKt;
  int nt = rest / numKt;
  int k = kt * 32 + ((lane & 16) ? 8 : 0) + (h < 8 ? h : h + 8);
  int c = nt * 16 + (lane & 15);
  P[idx] = f2bf(W[k * C + c]);
}

__global__ void cvt_bf16_kernel(const float* __restrict__ src,
                                __bf16* __restrict__ dst, int n) {
  int i = blockIdx.x * blockDim.x + threadIdx.x;
  if (i < n) dst[i] = f2bf(src[i]);
}

__global__ void init_state_kernel(const float* __restrict__ h0,
                                  __bf16* __restrict__ hbf,
                                  float* __restrict__ c, int n) {
  int i = blockIdx.x * blockDim.x + threadIdx.x;
  if (i < n) { hbf[i] = f2bf(h0[i]); c[i] = 0.0f; }
}

// xWx GEMM: out[M x 4H] = xbf[M x D] @ Wx + b.  Each wave computes a 32x64
// tile (2 M-tiles x 4 N-tiles, 8 accumulators): 6 fragment loads feed 8 WMMAs
// per K-step, and the fragment pipeline is double-buffered.
__global__ __launch_bounds__(256) void gemm_xwx_kernel(
    const __bf16* __restrict__ xbf,   // [M x DD] row-major bf16
    const __bf16* __restrict__ wxp,   // packed B fragments
    const float*  __restrict__ bias,  // [4H]
    float* __restrict__ outc) {       // [M x 4H] f32
  const int lane  = threadIdx.x & 31;
  const int wave  = threadIdx.x >> 5;
  const int strip = blockIdx.x * 8 + wave;   // 32768 strips total
  const int mt = strip >> 5;                 // 1024 row tiles of 32
  const int ns = strip & 31;                 // 32 col strips of 64
  const int m0 = mt * 32;
  const int n0 = ns * 64;
  const int numKt = DD / 32;                 // 16

  const __bf16* arow0 = xbf + (m0 + (lane & 15)) * DD + ((lane & 16) ? 8 : 0);
  const __bf16* arow1 = arow0 + 16 * DD;
  const int ntb = n0 >> 4;

  v8f acc[2][4] = {};

  // Prologue: stage kt = 0 fragments.
  v16bf aC[2], bC[4];
  aC[0] = load_a_frag(arow0);
  aC[1] = load_a_frag(arow1);
#pragma unroll
  for (int j = 0; j < 4; ++j)
    bC[j] = *(const v16bf*)(wxp + (((ntb + j) * numKt + 0) * 32 + lane) * 16);

#pragma unroll
  for (int kt = 0; kt < numKt; ++kt) {
    v16bf aN[2], bN[4];
    if (kt + 1 < numKt) {                    // issue next-stage loads first
      aN[0] = load_a_frag(arow0 + (kt + 1) * 32);
      aN[1] = load_a_frag(arow1 + (kt + 1) * 32);
#pragma unroll
      for (int j = 0; j < 4; ++j)
        bN[j] = *(const v16bf*)(wxp + (((ntb + j) * numKt + (kt + 1)) * 32 + lane) * 16);
    }
#pragma unroll
    for (int j = 0; j < 4; ++j) {
      acc[0][j] = __builtin_amdgcn_wmma_f32_16x16x32_bf16(
          false, aC[0], false, bC[j], (short)0, acc[0][j], false, false);
      acc[1][j] = __builtin_amdgcn_wmma_f32_16x16x32_bf16(
          false, aC[1], false, bC[j], (short)0, acc[1][j], false, false);
    }
    if (kt + 1 < numKt) {
      aC[0] = aN[0]; aC[1] = aN[1];
#pragma unroll
      for (int j = 0; j < 4; ++j) bC[j] = bN[j];
    }
  }

  const int rbase = (lane & 16) ? 8 : 0;   // C/D layout: lanes 16-31 hold M=8..15
  const int col   = lane & 15;
#pragma unroll
  for (int r = 0; r < 2; ++r) {
#pragma unroll
    for (int j = 0; j < 4; ++j) {
      const int cgl = n0 + j * 16 + col;
      const float bv = bias[cgl];
#pragma unroll
      for (int e = 0; e < 8; ++e) {
        outc[(m0 + r * 16 + rbase + e) * FOURH + cgl] = acc[r][j][e] + bv;
      }
    }
  }
}

// One recurrence step, fully fused: each wave owns a 16x16 (n, j) tile of h_t
// and computes the four gate accumulators (sharing the h_{t-1} A fragment),
// double-buffered over K, then sigmoid/tanh gates, c update, f32+bf16 stores.
__global__ __launch_bounds__(128) void lstm_step_kernel(
    const float*  __restrict__ xwx,   // [N*T x 4H] precomputed x@Wx + b
    const __bf16* __restrict__ whp,   // packed Wh fragments
    const __bf16* __restrict__ hin,   // [N x H] bf16 h_{t-1}
    __bf16* __restrict__ hout,        // [N x H] bf16 h_t
    float*  __restrict__ cst,         // [N x H] f32 cell state
    float*  __restrict__ out,         // [N x T x H] f32 output
    int t) {
  const int lane = threadIdx.x & 31;
  const int wave = threadIdx.x >> 5;
  const int tile = blockIdx.x * 4 + wave;    // 128 tiles: 4 n-tiles x 32 j-tiles
  const int n0 = (tile >> 5) * 16;
  const int jt = tile & 31;
  const int j0 = jt * 16;
  const int numKt = HH / 32;                 // 16

  const __bf16* arow = hin + (n0 + (lane & 15)) * HH + ((lane & 16) ? 8 : 0);

  v8f acc[4] = {};

  v16bf aC, bC[4];
  aC = load_a_frag(arow);
#pragma unroll
  for (int g = 0; g < 4; ++g)
    bC[g] = *(const v16bf*)(whp + (((g * 32 + jt) * numKt + 0) * 32 + lane) * 16);

#pragma unroll
  for (int kt = 0; kt < numKt; ++kt) {
    v16bf aN, bN[4];
    if (kt + 1 < numKt) {                    // prefetch next-stage fragments
      aN = load_a_frag(arow + (kt + 1) * 32);
#pragma unroll
      for (int g = 0; g < 4; ++g)
        bN[g] = *(const v16bf*)(whp + (((g * 32 + jt) * numKt + (kt + 1)) * 32 + lane) * 16);
    }
#pragma unroll
    for (int g = 0; g < 4; ++g) {
      acc[g] = __builtin_amdgcn_wmma_f32_16x16x32_bf16(
          false, aC, false, bC[g], (short)0, acc[g], false, false);
    }
    if (kt + 1 < numKt) {
      aC = aN;
#pragma unroll
      for (int g = 0; g < 4; ++g) bC[g] = bN[g];
    }
  }

  const int rbase = (lane & 16) ? 8 : 0;
  const int j = j0 + (lane & 15);
#pragma unroll
  for (int e = 0; e < 8; ++e) {
    const int n = n0 + rbase + e;
    const float* axr = xwx + (size_t)(n * TT + t) * FOURH;
    float pi = acc[0][e] + axr[0 * HH + j];
    float pf = acc[1][e] + axr[1 * HH + j];
    float po = acc[2][e] + axr[2 * HH + j];
    float pg = acc[3][e] + axr[3 * HH + j];
    float ig = 1.0f / (1.0f + __expf(-pi));
    float fg = 1.0f / (1.0f + __expf(-pf));
    float og = 1.0f / (1.0f + __expf(-po));
    float gx = fminf(fmaxf(pg, -15.0f), 15.0f);
    float eg = __expf(2.0f * gx);
    float gg = (eg - 1.0f) / (eg + 1.0f);            // tanh
    float cOld = cst[n * HH + j];
    float cNew = fg * cOld + ig * gg;
    cst[n * HH + j] = cNew;
    float cx = fminf(fmaxf(cNew, -15.0f), 15.0f);
    float ec = __expf(2.0f * cx);
    float th = (ec - 1.0f) / (ec + 1.0f);            // tanh(c)
    float h = og * th;
    out[((size_t)n * TT + t) * HH + j] = h;
    hout[n * HH + j] = f2bf(h);
  }
}

extern "C" void kernel_launch(void* const* d_in, const int* in_sizes, int n_in,
                              void* d_out, int out_size, void* d_ws, size_t ws_size,
                              hipStream_t stream) {
  const float* x  = (const float*)d_in[0];   // (N, T, D)
  const float* h0 = (const float*)d_in[1];   // (N, H)
  const float* Wx = (const float*)d_in[2];   // (D, 4H)
  const float* Wh = (const float*)d_in[3];   // (H, 4H)
  const float* b  = (const float*)d_in[4];   // (4H,)
  float* out = (float*)d_out;                // (N, T, H)

  char* ws = (char*)d_ws;
  size_t off = 0;
  float*  xwx = (float*)(ws + off);  off += (size_t)NB * TT * FOURH * 4;  // 256 MiB
  __bf16* xbf = (__bf16*)(ws + off); off += (size_t)NB * TT * DD * 2;     //  32 MiB
  __bf16* wxp = (__bf16*)(ws + off); off += (size_t)DD * FOURH * 2;       //   2 MiB
  __bf16* whp = (__bf16*)(ws + off); off += (size_t)HH * FOURH * 2;       //   2 MiB
  __bf16* hb0 = (__bf16*)(ws + off); off += (size_t)NB * HH * 2;
  __bf16* hb1 = (__bf16*)(ws + off); off += (size_t)NB * HH * 2;
  float*  cst = (float*)(ws + off);  off += (size_t)NB * HH * 4;
  __bf16* hb[2] = {hb0, hb1};

  // Prep: pack weights to WMMA fragment order, convert x, init h0/c.
  {
    const int nw = DD * FOURH;
    pack_b_kernel<<<(nw + 255) / 256, 256, 0, stream>>>(Wx, wxp, DD, FOURH);
    pack_b_kernel<<<(nw + 255) / 256, 256, 0, stream>>>(Wh, whp, HH, FOURH);
  }
  {
    const int nx = NB * TT * DD;
    cvt_bf16_kernel<<<(nx + 255) / 256, 256, 0, stream>>>(x, xbf, nx);
  }
  {
    const int nh = NB * HH;
    init_state_kernel<<<(nh + 255) / 256, 256, 0, stream>>>(h0, hb[0], cst, nh);
  }

  // Phase 1: xWx = x @ Wx + b  (32768 waves, one 32x64 tile each)
  gemm_xwx_kernel<<<4096, 256, 0, stream>>>(xbf, wxp, b, xwx);

  // Phase 2: 512 serialized fused steps (h ping-pong in bf16).
  for (int t = 0; t < TT; ++t) {
    lstm_step_kernel<<<32, 128, 0, stream>>>(
        xwx, whp, hb[t & 1], hb[(t + 1) & 1], cst, out, t);
  }
}
